// SwinTransformerDecoder_69758858822046
// MI455X (gfx1250) — compile-verified, matrix operand order checked by reading, not run on hardware
//
#include <hip/hip_runtime.h>

// ---------------------------------------------------------------------------
// Swin shifted-window attention for MI455X (gfx1250).
//   B=16, H=W=128, C=256, window 8x8 (N=64), shift 4, heads=8, hd=32.
//   4096 independent windows -> ONE fused workgroup per window:
//     P0 gather x (roll -4) -> LDS bf16
//     P1 QKV GEMM 64x256x768 -> q,k,vT kept in LDS
//     P2 attention: S^T = K@Q^T, softmax (1x shfl_xor16), O^T = V^T@P^T
//     P3 proj GEMM 64x256x256 -> window-reverse + roll(+4) scatter to HBM
//   Only x (268MB) is read and out (268MB) written to HBM; q/k/v/X2 stay in
//   the 320KB WGP LDS (~156KB/workgroup, 2 WGs/WGP).  All GEMMs use
//   v_wmma_f32_16x16x32_bf16 with f32 accumulation.
// ---------------------------------------------------------------------------

typedef __attribute__((ext_vector_type(16))) __bf16 bf16x16;
typedef __attribute__((ext_vector_type(8)))  float  v8f;

union FragU { bf16x16 v; uint4 q[2]; unsigned short u[16]; };
union H8   { uint4 q; unsigned short u[8]; };

__device__ __forceinline__ unsigned short f2bf(float f) {
  unsigned u = __builtin_bit_cast(unsigned, f);   // f32->bf16 RNE
  return (unsigned short)((u + 0x7FFFu + ((u >> 16) & 1u)) >> 16);
}

__device__ __forceinline__ v8f wmma_bf16(bf16x16 a, bf16x16 b, v8f c) {
  return __builtin_amdgcn_wmma_f32_16x16x32_bf16(
      false, a, false, b, (short)0, c, false, false);
}

// ---- LDS layout (elements of ushort unless noted) -------------------------
//  sX  [64][264]      16896  (x tile, reused as X2 tile in P2/P3)
//  sQ  [8][64][40]    20480  (row pad->16B-aligned, conflict-free b128 reads)
//  sK  [8][64][40]    20480
//  sVT [8][32][72]    18432
//  sTab[225*8] f32    (byte offset 152576)
#define LDS_BYTES 159776

// ---------------------------------------------------------------------------
// Kernel 0: weights f32 -> bf16 (row-major [outC][inC] = B-fragment order).
// ---------------------------------------------------------------------------
__global__ __launch_bounds__(256)
void swin_wconv(const float* __restrict__ qkvW, const float* __restrict__ projW,
                unsigned short* __restrict__ wq, unsigned short* __restrict__ wp) {
  int i = blockIdx.x * 256 + threadIdx.x;       // grid covers 768*256
  if (i < 768 * 256) wq[i] = f2bf(qkvW[i]);
  if (i < 256 * 256) wp[i] = f2bf(projW[i]);
}

// ---------------------------------------------------------------------------
// Fused per-window kernel: 4096 blocks x 256 threads (8 waves).
// ---------------------------------------------------------------------------
__global__ __launch_bounds__(256)
void swin_fused(const float* __restrict__ x,
                const unsigned short* __restrict__ wq,
                const float* __restrict__ qkv_bias,
                const unsigned short* __restrict__ wp,
                const float* __restrict__ proj_bias,
                const float* __restrict__ relTab,
                float* __restrict__ out) {
  extern __shared__ char smem_raw[];
  unsigned short* sX  = (unsigned short*)smem_raw;   // [64][264], later X2
  unsigned short* sQ  = sX + 16896;                  // [h][m][40]
  unsigned short* sK  = sQ + 20480;                  // [h][m][40]
  unsigned short* sVT = sK + 20480;                  // [h][d][72]
  float*          sTab = (float*)(smem_raw + 152576);

  const int w  = blockIdx.x;
  const int b  = w >> 8, wim = w & 255, wy = wim >> 4, wx = wim & 15;
  const int tid = threadIdx.x;
  const int wv = tid >> 5, lane = tid & 31;
  const int l15 = lane & 15, hb = lane >> 4;         // half-wave 0/1
  const int koff = hb ? 8 : 0;                       // ISA A/B K-interleave

  // ---------------- P0: gather shifted window -> LDS bf16; table -> LDS ----
  for (int i = tid; i < 64 * 64; i += 256) {
    const int t  = i >> 6, c4 = (i & 63) << 2;
    const int ty = t >> 3, tx = t & 7;
    const int hs  = (wy * 8 + ty + 4) & 127;         // roll(-4)
    const int ws2 = (wx * 8 + tx + 4) & 127;
    const float4 f = *(const float4*)(x + (((long)b * 128 + hs) * 128 + ws2) * 256 + c4);
    ushort4 sv;
    sv.x = f2bf(f.x); sv.y = f2bf(f.y); sv.z = f2bf(f.z); sv.w = f2bf(f.w);
    *(ushort4*)&sX[t * 264 + c4] = sv;
  }
  for (int i = tid; i < 225 * 8; i += 256) sTab[i] = relTab[i];
  __syncthreads();

  // ---------------- P1: QKV GEMM 64x256x768, epilogue into LDS -------------
  for (int ntl = 0; ntl < 6; ++ntl) {                // 8 waves x 6 = 48 N-tiles
    const int nt  = wv * 6 + ntl;
    const int col = nt * 16 + l15;
    const float bias = qkv_bias[col];
    if (ntl < 5)                                      // prefetch next tile row
      __builtin_prefetch(wq + (col + 16) * 256, 0, 3);

    v8f acc[4] = {};
    #pragma unroll
    for (int k = 0; k < 8; ++k) {                    // K = 256
      const int k0 = k * 32;
      FragU bw;
      const unsigned short* wr = wq + col * 256 + k0 + koff;
      bw.q[0] = *(const uint4*)(wr);
      bw.q[1] = *(const uint4*)(wr + 16);
      #pragma unroll
      for (int mt = 0; mt < 4; ++mt) {
        FragU a;
        const unsigned short* ar = &sX[(mt * 16 + l15) * 264 + k0 + koff];
        a.q[0] = *(const uint4*)(ar);
        a.q[1] = *(const uint4*)(ar + 16);
        acc[mt] = wmma_bf16(a.v, bw.v, acc[mt]);
      }
    }
    const int which = col >> 8;                      // 0=q 1=k 2=v (tile-uniform)
    const int c2 = col & 255;
    const int head = c2 >> 5, d = c2 & 31;
    const float scale = (which == 0) ? 0.17677669529663687f : 1.0f;
    unsigned short* dst = (which == 0) ? sQ : (which == 1) ? sK : sVT;
    #pragma unroll
    for (int mt = 0; mt < 4; ++mt)
      #pragma unroll
      for (int r = 0; r < 8; ++r) {
        const int m = mt * 16 + hb * 8 + r;          // token
        const float val = (acc[mt][r] + bias) * scale;
        const int off = (which == 2) ? (head * 2304 + d * 72 + m)   // vT [d][m]
                                     : (head * 2560 + m * 40 + d);  // q,k [m][d]
        dst[off] = f2bf(val);
      }
  }
  __syncthreads();

  // ---------------- P2: attention, one wave per head -----------------------
  {
    const int h = wv;
    const unsigned short* qh = sQ + h * 2560;
    const unsigned short* kh = sK + h * 2560;
    const unsigned short* vh = sVT + h * 2304;

    FragU aK[4], aV[2][2];
    #pragma unroll
    for (int t = 0; t < 4; ++t) {                    // K rows (A operand)
      const unsigned short* kr = kh + (t * 16 + l15) * 40 + koff;
      aK[t].q[0] = *(const uint4*)(kr); aK[t].q[1] = *(const uint4*)(kr + 16);
    }
    #pragma unroll
    for (int dt = 0; dt < 2; ++dt)                   // V^T rows (A operand)
      #pragma unroll
      for (int kt = 0; kt < 2; ++kt) {
        const unsigned short* vr = vh + (dt * 16 + l15) * 72 + kt * 32 + koff;
        aV[dt][kt].q[0] = *(const uint4*)(vr); aV[dt][kt].q[1] = *(const uint4*)(vr + 16);
      }

    const int regx_on = (wx == 15), regy_on = (wy == 15);

    #pragma unroll
    for (int mt = 0; mt < 4; ++mt) {                 // per query tile (low regs)
      // S^T column block mt: 4 WMMAs
      FragU bQ;
      const unsigned short* qr = qh + (mt * 16 + l15) * 40 + koff;
      bQ.q[0] = *(const uint4*)(qr); bQ.q[1] = *(const uint4*)(qr + 16);
      v8f sT[4];
      #pragma unroll
      for (int nt = 0; nt < 4; ++nt) {
        v8f z = {};
        sT[nt] = wmma_bf16(aK[nt].v, bQ.v, z);
      }

      // bias + shift mask + softmax over n (lane owns query column m)
      const int m = mt * 16 + l15;
      const int my = m >> 3, mxc = m & 7;
      const int regm = (regy_on ? ((my < 4) ? 1 : 2) : 0) * 3
                     + (regx_on ? ((mxc < 4) ? 1 : 2) : 0);
      float mx = -3.0e38f;
      #pragma unroll
      for (int nt = 0; nt < 4; ++nt)
        #pragma unroll
        for (int r = 0; r < 8; ++r) {
          const int n = nt * 16 + hb * 8 + r;
          const int ny = n >> 3, nxc = n & 7;
          const int regn = (regy_on ? ((ny < 4) ? 1 : 2) : 0) * 3
                         + (regx_on ? ((nxc < 4) ? 1 : 2) : 0);
          const int idx = (my - ny + 7) * 15 + (mxc - nxc + 7);
          float v = sT[nt][r] + sTab[idx * 8 + h];
          if (regm != regn) v -= 100.0f;
          sT[nt][r] = v;
          mx = fmaxf(mx, v);
        }
      mx = fmaxf(mx, __shfl_xor(mx, 16, 32));        // partner holds other 32 n
      float s = 0.f;
      #pragma unroll
      for (int nt = 0; nt < 4; ++nt)
        #pragma unroll
        for (int r = 0; r < 8; ++r) {
          const float e = __expf(sT[nt][r] - mx);
          sT[nt][r] = e; s += e;
        }
      s += __shfl_xor(s, 16, 32);
      const float inv = 1.0f / s;

      // P^T registers are already in B-fragment layout: per-lane pack only
      FragU bP[2];
      #pragma unroll
      for (int kt = 0; kt < 2; ++kt)
        #pragma unroll
        for (int j = 0; j < 8; ++j) {
          bP[kt].u[j]     = f2bf(sT[2 * kt][j] * inv);
          bP[kt].u[8 + j] = f2bf(sT[2 * kt + 1][j] * inv);
        }

      // O^T block: 4 WMMAs; store into X2 tile (reuses sX, dead after P1)
      #pragma unroll
      for (int dt = 0; dt < 2; ++dt) {
        v8f z = {};
        z = wmma_bf16(aV[dt][0].v, bP[0].v, z);
        z = wmma_bf16(aV[dt][1].v, bP[1].v, z);
        const int dbase = dt * 16 + hb * 8;          // 8 consecutive d per lane
        H8 h8;
        #pragma unroll
        for (int r = 0; r < 8; ++r) h8.u[r] = f2bf(z[r]);
        *(uint4*)&sX[m * 264 + h * 32 + dbase] = h8.q;
      }
    }
  }
  __syncthreads();

  // ---------------- P3: proj GEMM 64x256x256 + scatter ---------------------
  {
    const int mt = wv & 3, ng = wv >> 2;             // wave -> (M tile, N half)
    v8f acc[8] = {};
    const unsigned short* arow = sX + (mt * 16 + l15) * 264;
    #pragma unroll
    for (int k = 0; k < 8; ++k) {                    // K = 256
      const int k0 = k * 32;
      FragU a;
      a.q[0] = *(const uint4*)(arow + k0 + koff);
      a.q[1] = *(const uint4*)(arow + k0 + koff + 16);
      #pragma unroll
      for (int n2 = 0; n2 < 8; ++n2) {
        const int nrow = (ng * 8 + n2) * 16 + l15;
        FragU bw;
        const unsigned short* br = wp + nrow * 256 + k0 + koff;
        bw.q[0] = *(const uint4*)(br);
        bw.q[1] = *(const uint4*)(br + 16);
        acc[n2] = wmma_bf16(a.v, bw.v, acc[n2]);
      }
    }
    #pragma unroll
    for (int n2 = 0; n2 < 8; ++n2) {
      const int c = (ng * 8 + n2) * 16 + l15;
      const float pb = proj_bias[c];
      #pragma unroll
      for (int r = 0; r < 8; ++r) {
        const int m  = mt * 16 + hb * 8 + r;
        const int ty = m >> 3, tx = m & 7;
        const int hg = (wy * 8 + ty + 4) & 127;      // window-reverse + roll(+4)
        const int wg = (wx * 8 + tx + 4) & 127;
        out[(((long)b * 128 + hg) * 128 + wg) * 256 + c] = acc[n2][r] + pb;
      }
    }
  }
}

// ---------------------------------------------------------------------------
// Launch.  Workspace: [0,393216) qkv_weight bf16; [393216,524288) proj bf16.
// ---------------------------------------------------------------------------
extern "C" void kernel_launch(void* const* d_in, const int* in_sizes, int n_in,
                              void* d_out, int out_size, void* d_ws, size_t ws_size,
                              hipStream_t stream) {
  (void)in_sizes; (void)n_in; (void)out_size; (void)ws_size;
  const float* x     = (const float*)d_in[0];
  const float* qkvW  = (const float*)d_in[1];
  const float* qkvB  = (const float*)d_in[2];
  const float* projW = (const float*)d_in[3];
  const float* projB = (const float*)d_in[4];
  const float* relT  = (const float*)d_in[5];

  char* ws = (char*)d_ws;
  unsigned short* wqb = (unsigned short*)(ws);
  unsigned short* wpb = (unsigned short*)(ws + 393216);

  hipLaunchKernelGGL(swin_wconv, dim3(768),  dim3(256), 0, stream, qkvW, projW, wqb, wpb);
  hipLaunchKernelGGL(swin_fused, dim3(4096), dim3(256), LDS_BYTES, stream,
                     x, wqb, qkvB, wpb, projB, relT, (float*)d_out);
}
